// W4A16Linear_34153579938182
// MI455X (gfx1250) — compile-verified, matrix operand order checked by reading, not run on hardware
//
#include <hip/hip_runtime.h>

// ---------------------------------------------------------------------------
// W4A16 group-quantized linear:  y = x @ dequant(W_q)^T + bias   (MI455X)
//   x [M,K] bf16 | W_q [N,K] int32 codes | scales/zeros [N,K/128] bf16
//   bias [N] bf16 | out [M,N] bf16
// Compute-bound => keep math on v_wmma_f32_16x16x32_bf16; dequant once per
// (N-tile, K-slice) into LDS with fma (q*s - z*s), amortized over 128 M rows.
// BK=64: one scale/zero per step, 16 WMMAs per wave per barrier pair.
// ---------------------------------------------------------------------------

typedef __bf16 bf16;
typedef __bf16 bf16x8   __attribute__((ext_vector_type(8)));
typedef __bf16 bf16x16  __attribute__((ext_vector_type(16)));
typedef float  f32x8    __attribute__((ext_vector_type(8)));
typedef int    i32x4    __attribute__((ext_vector_type(4)));

#define MDIM 4096
#define KDIM 4096
#define NDIM 11008
#define GSZ  128
#define NGRP (KDIM / GSZ)   // 32

#define BM   128            // block tile M
#define BN   128            // block tile N
#define BK   64             // K slice per staging round (2 WMMA K-steps)
#define LDSS 72             // padded LDS row stride (elements): 144B, 16B-aligned,
                            // conflict-free for 16-lane fragment reads

__global__ __launch_bounds__(256)
void w4a16_wmma_bf16_kernel(const bf16* __restrict__ x,
                            const int*  __restrict__ wq,
                            const bf16* __restrict__ scales,
                            const bf16* __restrict__ zeros,
                            const bf16* __restrict__ bias,
                            bf16*       __restrict__ out)
{
    __shared__ bf16 xs[BM * LDSS];  // x tile [m][k]
    __shared__ bf16 ws[BN * LDSS];  // dequantized W tile [n][k]

    const int tid  = threadIdx.x;
    const int lane = tid & 31;      // wave32
    const int wid  = tid >> 5;      // 8 waves

    const int blockN = blockIdx.x * BN;
    const int blockM = blockIdx.y * BM;

    // 8 waves arranged 4 (M) x 2 (N); each wave owns a 32x64 output region
    const int waveM = (wid >> 1) * 32;  // 0,32,64,96
    const int waveN = (wid & 1) * 64;   // 0,64

    f32x8 acc[2][4];
#pragma unroll
    for (int i = 0; i < 2; ++i)
#pragma unroll
        for (int j = 0; j < 4; ++j)
            acc[i][j] = f32x8{};

    // staging: each thread owns 32 elements of one row of each tile
    const int sRow = tid >> 1;          // 0..127
    const int sK   = (tid & 1) * 32;    // 0 or 32

    const bf16* xg = x  + (size_t)(blockM + sRow) * KDIM + sK;
    const int*  wg = wq + (size_t)(blockN + sRow) * KDIM + sK;
    const bf16* sg = scales + (size_t)(blockN + sRow) * NGRP;
    const bf16* zg = zeros  + (size_t)(blockN + sRow) * NGRP;

    for (int k0 = 0; k0 < KDIM; k0 += BK) {
        // ---- global loads for this K slice ----
        bf16x8 xv[4];
#pragma unroll
        for (int i = 0; i < 4; ++i)
            xv[i] = *(const bf16x8*)(xg + k0 + 8 * i);

        i32x4 qv[8];
#pragma unroll
        for (int i = 0; i < 8; ++i)
            qv[i] = *(const i32x4*)(wg + k0 + 4 * i);

        if (k0 + BK < KDIM) {   // next slice into cache (global_prefetch_b8)
            __builtin_prefetch(xg + k0 + BK, 0, 1);
            __builtin_prefetch(wg + k0 + BK, 0, 1);
        }

        // ---- dequant: w = q*s + (-z*s), one group per 64-wide slice ----
        const int   g  = k0 >> 7;               // k0 / GSZ
        const float s  = (float)sg[g];
        const float nz = -(float)zg[g] * s;

        bf16x8 wv[4];
#pragma unroll
        for (int i = 0; i < 8; ++i)
#pragma unroll
            for (int j = 0; j < 4; ++j)
                wv[i >> 1][(i & 1) * 4 + j] =
                    (bf16)__builtin_fmaf((float)qv[i][j], s, nz);

        __syncthreads();  // previous iteration's readers done
#pragma unroll
        for (int i = 0; i < 4; ++i) {
            *(bf16x8*)&xs[sRow * LDSS + sK + 8 * i] = xv[i];
            *(bf16x8*)&ws[sRow * LDSS + sK + 8 * i] = wv[i];
        }
        __syncthreads();  // tiles visible to all waves

        // ---- 2 WMMA K-steps per staging round ----
        // 16-bit A/B layout: lane 0-15 takes K=base+0..7 (lo), K=base+16..23 (hi);
        // lane 16-31 takes K=base+8..15 / K=base+24..31
        const int l15 = lane & 15;
        const int kbb = (lane < 16) ? 0 : 8;

#pragma unroll
        for (int ks = 0; ks < 2; ++ks) {
            const int kb = ks * 32 + kbb;

            bf16x16 afrag[2];
#pragma unroll
            for (int tm = 0; tm < 2; ++tm) {
                const int row = waveM + tm * 16 + l15;
                bf16x8 lo = *(const bf16x8*)&xs[row * LDSS + kb];
                bf16x8 hi = *(const bf16x8*)&xs[row * LDSS + kb + 16];
                afrag[tm] = __builtin_shufflevector(lo, hi,
                    0,1,2,3,4,5,6,7,8,9,10,11,12,13,14,15);
            }
            bf16x16 bfrag[4];
#pragma unroll
            for (int tn = 0; tn < 4; ++tn) {
                const int col = waveN + tn * 16 + l15;
                bf16x8 lo = *(const bf16x8*)&ws[col * LDSS + kb];
                bf16x8 hi = *(const bf16x8*)&ws[col * LDSS + kb + 16];
                bfrag[tn] = __builtin_shufflevector(lo, hi,
                    0,1,2,3,4,5,6,7,8,9,10,11,12,13,14,15);
            }
#pragma unroll
            for (int tm = 0; tm < 2; ++tm)
#pragma unroll
                for (int tn = 0; tn < 4; ++tn)
                    acc[tm][tn] = __builtin_amdgcn_wmma_f32_16x16x32_bf16(
                        /*neg_a=*/false, afrag[tm],
                        /*neg_b=*/false, bfrag[tn],
                        /*c_mod=*/(short)0, acc[tm][tn],
                        /*reuse_a=*/false, /*reuse_b=*/false);
        }
    }

    // ---- epilogue: bias add, f32 -> bf16, store ----
    // C/D layout: VGPR r -> M = r (lanes 0-15) or 8+r (lanes 16-31); N = lane%16
    const int l15   = lane & 15;
    const int rbase = (lane < 16) ? 0 : 8;
#pragma unroll
    for (int tn = 0; tn < 4; ++tn) {
        const int col = blockN + waveN + tn * 16 + l15;
        const float bv = (float)bias[col];
#pragma unroll
        for (int tm = 0; tm < 2; ++tm) {
            const int row0 = blockM + waveM + tm * 16 + rbase;
#pragma unroll
            for (int r = 0; r < 8; ++r) {
                out[(size_t)(row0 + r) * NDIM + col] = (bf16)(acc[tm][tn][r] + bv);
            }
        }
    }
}

extern "C" void kernel_launch(void* const* d_in, const int* in_sizes, int n_in,
                              void* d_out, int out_size, void* d_ws, size_t ws_size,
                              hipStream_t stream) {
    (void)in_sizes; (void)n_in; (void)out_size; (void)d_ws; (void)ws_size;
    const bf16* x    = (const bf16*)d_in[0];
    const int*  wqp  = (const int*)d_in[1];
    const bf16* sc   = (const bf16*)d_in[2];
    const bf16* zp   = (const bf16*)d_in[3];
    const bf16* bias = (const bf16*)d_in[4];
    bf16*       out  = (bf16*)d_out;

    dim3 grid(NDIM / BN, MDIM / BM);  // (86, 32)
    dim3 block(256);
    w4a16_wmma_bf16_kernel<<<grid, block, 0, stream>>>(x, wqp, sc, zp, bias, out);
}
